// TransFusionHeadAnchorMatching_29025388987062
// MI455X (gfx1250) — matrix operand order repaired and checked by reading, not compiled, not run on hardware
//
#include <hip/hip_runtime.h>
#include <hip/hip_bf16.h>
#include <math.h>

#define BN   2
#define CINX 512
#define HH   180
#define WW   180
#define HWD  (HH * WW)
#define HID  128
#define TD   60
#define NC   10
#define NP   200
#define NHD  8
#define HD   16
#define FFN  256

typedef __attribute__((ext_vector_type(16))) __bf16 bf16x16;
typedef __attribute__((ext_vector_type(8)))  float  f32x8;

// ---------------------------------------------------------------------------
// CDNA5 async copy: global -> LDS DMA (ASYNCcnt tracked), ISA ch.10 opcode 96.
// ---------------------------------------------------------------------------
__device__ __forceinline__ void async_load_f32_to_lds(void* lds_ptr, const float* gptr)
{
    unsigned lds_off = (unsigned)(uintptr_t)lds_ptr;          // flat addr[31:0] == LDS offset
    unsigned long long ga = (unsigned long long)(uintptr_t)gptr;
    asm volatile("global_load_async_to_lds_b32 %0, %1, off"
                 :: "v"(lds_off), "v"(ga) : "memory");
}
__device__ __forceinline__ void wait_async()
{
    asm volatile("s_wait_asynccnt 0x0" ::: "memory");
}

// ---------------------------------------------------------------------------
// Implicit-GEMM 3x3 conv, bf16 WMMA, f32 accumulate.
// Block = 128 (4 waves). Output tile per block: 64 Cout x 32 pixels.
// Per wave: 16 Cout x 32 px; K unrolled by 64 -> 4 v_wmma per barrier.
// Weight tile staged raw fp32 via async-to-LDS; im2col tile staged as bf16.
// grid.x = B * H * ceil(W/32); grid.y = ceil(Cout/64)
// mode 0: out = conv + bias
// mode 1: out = relu(conv * gam + bet)
// mode 2: out = conv * exp(msc[0]) + mbias[0]
// ---------------------------------------------------------------------------
__global__ __launch_bounds__(128)
void conv3x3_wmma(const float* __restrict__ x, const float* __restrict__ wgt,
                  const float* __restrict__ bias, const float* __restrict__ gam,
                  const float* __restrict__ bet, const float* __restrict__ msc,
                  const float* __restrict__ mbias, float* __restrict__ out,
                  int Cin, int Cout, int mode)
{
    __shared__ float  As[64][65];   // weight tile [cout-in-block][k0..63] (fp32, async-staged)
    __shared__ __bf16 Bs[64][34];   // im2col tile [k][pixel0..31] (bf16, converted once)

    const int wtiles = (WW + 31) >> 5;
    const int bx = blockIdx.x;
    const int wt = bx % wtiles;
    const int hh = (bx / wtiles) % HH;
    const int b  = bx / (wtiles * HH);
    const int w0 = wt * 32;
    const int coBlk = blockIdx.y * 64;
    const int tid  = threadIdx.x;
    const int lane = tid & 31;
    const int wave = tid >> 5;
    const int Ktot = Cin * 9;
    const bool interior = (hh > 0) && (hh < HH - 1) && (w0 > 0) && (w0 + 32 < WW);

    f32x8 acc0 = {}, acc1 = {};

    for (int kk = 0; kk < Ktot; kk += 64) {
        const bool kfull = (kk + 64) <= Ktot;
        // ---- stage weight tile (64x64) raw fp32 via async DMA ----
        #pragma unroll
        for (int i = 0; i < 32; ++i) {
            int idx = tid + i * 128;
            int row = idx >> 6, kl = idx & 63;
            int k = kk + kl, co = coBlk + row;
            if (co < Cout && k < Ktot)
                async_load_f32_to_lds(&As[row][kl], &wgt[(size_t)co * Ktot + k]);
            else
                As[row][kl] = 0.f;
        }
        // ---- stage im2col tile (64x32) as bf16 ----
        if (interior && kfull) {
            #pragma unroll
            for (int i = 0; i < 16; ++i) {
                int idx = tid + i * 128;
                int kl = idx >> 5, n = idx & 31;
                int k = kk + kl;
                int ci = k / 9, r = k % 9;
                int hy = hh + r / 3 - 1;
                int wx = w0 + n + r % 3 - 1;
                Bs[kl][n] = (__bf16)x[(((size_t)b * Cin + ci) * HH + hy) * WW + wx];
            }
        } else {
            #pragma unroll
            for (int i = 0; i < 16; ++i) {
                int idx = tid + i * 128;
                int kl = idx >> 5, n = idx & 31;
                int k = kk + kl;
                float v = 0.f;
                if (k < Ktot) {
                    int ci = k / 9, r = k % 9;
                    int hy = hh + r / 3 - 1;
                    int wx = w0 + n + r % 3 - 1;
                    if (hy >= 0 && hy < HH && wx >= 0 && wx < WW)
                        v = x[(((size_t)b * Cin + ci) * HH + hy) * WW + wx];
                }
                Bs[kl][n] = (__bf16)v;
            }
        }
        wait_async();
        __syncthreads();

        // ---- 2 K-subtiles x 2 N-subtiles = 4 WMMAs, A-fragment reused ----
        const int mrow  = wave * 16 + (lane & 15);
        const int khalf = (lane >> 4) << 3;
        const int krow  = (lane >> 4) << 4;
        const int nlo   = lane & 15;
        #pragma unroll
        for (int ks = 0; ks < 64; ks += 32) {
            bf16x16 af, b0, b1;
            #pragma unroll
            for (int i = 0; i < 16; ++i) {
                af[i] = (__bf16)As[mrow][ks + krow + i];
                b0[i] = Bs[ks + krow + i][nlo];
                b1[i] = Bs[ks + krow + i][16 + nlo];
            }
            acc0 = __builtin_amdgcn_wmma_f32_16x16x32_bf16(false, af, false, b0,
                                                           (short)0, acc0, false, false);
            acc1 = __builtin_amdgcn_wmma_f32_16x16x32_bf16(false, af, false, b1,
                                                           (short)0, acc1, false, false);
        }
        __syncthreads();
    }

    // ---- epilogue: C layout VGPR r -> M = r + 8*(lane>=16), N = lane&15 ----
    const int n = lane & 15;
    #pragma unroll
    for (int r = 0; r < 8; ++r) {
        int m  = r + ((lane >> 4) << 3);
        int co = coBlk + wave * 16 + m;
        if (co < Cout) {
            float e0 = acc0[r], e1 = acc1[r];
            if (mode == 0)      { e0 += bias[co]; e1 += bias[co]; }
            else if (mode == 1) {
                e0 = e0 * gam[co] + bet[co]; e0 = e0 > 0.f ? e0 : 0.f;
                e1 = e1 * gam[co] + bet[co]; e1 = e1 > 0.f ? e1 : 0.f;
            } else {
                float sc = __expf(msc[0]), bb = mbias[0];
                e0 = e0 * sc + bb; e1 = e1 * sc + bb;
            }
            size_t rowb = (((size_t)b * Cout + co) * HH + hh) * WW;
            if (w0 + n      < WW) out[rowb + w0 + n]      = e0;
            if (w0 + 16 + n < WW) out[rowb + w0 + 16 + n] = e1;
        }
    }
}

// ---------------------------------------------------------------------------
// C[M,N] = A[M,K] * W[N,K]^T + bias[N]   (bf16 WMMA, f32 accum)
// Block = 128 (4 waves). Tile per block: 32 M x 64 N; K unrolled by 64.
// Per wave: 16 N x 32 M -> 4 v_wmma per barrier (A-fragment reused).
// Weight tile staged raw fp32 via async-to-LDS; activation tile as bf16.
// grid.x = ceil(M/32); grid.y = ceil(N/64)
// ---------------------------------------------------------------------------
__global__ __launch_bounds__(128)
void gemm_nt_wmma(const float* __restrict__ A, const float* __restrict__ Wt,
                  const float* __restrict__ bias, float* __restrict__ C,
                  int M, int N, int K, int relu)
{
    __shared__ __bf16 Ash[32][66];  // activations [m][k0..63] (bf16, shared by all waves)
    __shared__ float  Wsh[64][65];  // weights [n][k0..63] (fp32, async-staged)
    const int mBlk = blockIdx.x * 32;
    const int nBlk = blockIdx.y * 64;
    const int tid  = threadIdx.x;
    const int lane = tid & 31;
    const int wave = tid >> 5;

    f32x8 acc0 = {}, acc1 = {};
    for (int kk = 0; kk < K; kk += 64) {
        #pragma unroll
        for (int i = 0; i < 16; ++i) {
            int idx = tid + i * 128;
            int row = idx >> 6, kl = idx & 63;
            int m = mBlk + row, k = kk + kl;
            Ash[row][kl] = (__bf16)((m < M && k < K) ? A[(size_t)m * K + k] : 0.f);
        }
        #pragma unroll
        for (int i = 0; i < 32; ++i) {
            int idx = tid + i * 128;
            int row = idx >> 6, kl = idx & 63;
            int nn = nBlk + row, k = kk + kl;
            if (nn < N && k < K)
                async_load_f32_to_lds(&Wsh[row][kl], &Wt[(size_t)nn * K + k]);
            else
                Wsh[row][kl] = 0.f;
        }
        wait_async();
        __syncthreads();

        const int khalf = (lane >> 4) << 3;
        const int kcol  = (lane >> 4) << 4;
        const int nlo   = lane & 15;
        #pragma unroll
        for (int ks = 0; ks < 64; ks += 32) {
            bf16x16 af, b0, b1;
            #pragma unroll
            for (int i = 0; i < 16; ++i) {
                af[i] = (__bf16)Wsh[wave * 16 + nlo][ks + kcol + i];
                b0[i] = Ash[nlo][ks + kcol + i];
                b1[i] = Ash[16 + nlo][ks + kcol + i];
            }
            acc0 = __builtin_amdgcn_wmma_f32_16x16x32_bf16(false, af, false, b0,
                                                           (short)0, acc0, false, false);
            acc1 = __builtin_amdgcn_wmma_f32_16x16x32_bf16(false, af, false, b1,
                                                           (short)0, acc1, false, false);
        }
        __syncthreads();
    }
    const int m0 = mBlk + (lane & 15);
    const int m1 = m0 + 16;
    #pragma unroll
    for (int r = 0; r < 8; ++r) {
        int nn = nBlk + wave * 16 + r + ((lane >> 4) << 3);
        if (nn < N) {
            float v0 = acc0[r] + bias[nn];
            float v1 = acc1[r] + bias[nn];
            if (relu) { v0 = v0 > 0.f ? v0 : 0.f; v1 = v1 > 0.f ? v1 : 0.f; }
            if (m0 < M) C[(size_t)m0 * N + nn] = v0;
            if (m1 < M) C[(size_t)m1 * N + nn] = v1;
        }
    }
}

// ---------------------------------------------------------------------------
// dynamic anchor kernels: wdyn = normalize(relu(av@aw1^T+ab1)@aw2^T+ab2)
// ---------------------------------------------------------------------------
__global__ __launch_bounds__(256)
void wdyn_kernel(const float* __restrict__ av, const float* __restrict__ aw1,
                 const float* __restrict__ ab1, const float* __restrict__ aw2,
                 const float* __restrict__ ab2, float* __restrict__ wdyn)
{
    __shared__ float t1[NC * TD];
    __shared__ float raw[NC * TD * 9];
    const int tid = threadIdx.x;
    for (int idx = tid; idx < NC * TD; idx += 256) {
        int c = idx / TD, j = idx % TD;
        float s = ab1[j];
        for (int k = 0; k < TD; ++k) s += av[c * TD + k] * aw1[j * TD + k];
        t1[idx] = s > 0.f ? s : 0.f;
    }
    __syncthreads();
    for (int idx = tid; idx < NC * TD * 9; idx += 256) {
        int c = idx / (TD * 9), j = idx % (TD * 9);
        float s = ab2[j];
        for (int k = 0; k < TD; ++k) s += t1[c * TD + k] * aw2[j * TD + k];
        raw[idx] = s;
    }
    __syncthreads();
    for (int idx = tid; idx < NC * 9; idx += 256) {
        int c = idx / 9, r = idx % 9;
        float nrm = 0.f;
        for (int t = 0; t < TD; ++t) { float v = raw[c * 540 + t * 9 + r]; nrm += v * v; }
        nrm = 1.f / (1e-8f + sqrtf(nrm));
        for (int t = 0; t < TD; ++t)
            wdyn[c * 540 + t * 9 + r] = raw[c * 540 + t * 9 + r] * nrm;
    }
}

// per-pixel L2 norm over channel axis
__global__ void channel_norm_kernel(const float* __restrict__ in,
                                    float* __restrict__ outp, int C, int n)
{
    int i = blockIdx.x * blockDim.x + threadIdx.x;
    if (i >= n) return;
    int b = i / HWD, s = i % HWD;
    const float* p = in + ((size_t)b * C) * HWD + s;
    float nrm = 0.f;
    for (int c = 0; c < C; ++c) { float v = p[(size_t)c * HWD]; nrm += v * v; }
    nrm = 1.f / (1e-8f + sqrtf(nrm));
    float* q = outp + ((size_t)b * C) * HWD + s;
    for (int c = 0; c < C; ++c) q[(size_t)c * HWD] = p[(size_t)c * HWD] * nrm;
}

__global__ void sigmoid_kernel(const float* __restrict__ in, float* __restrict__ outp, int n)
{
    int i = blockIdx.x * blockDim.x + threadIdx.x;
    if (i >= n) return;
    outp[i] = 1.f / (1.f + __expf(-in[i]));
}

// 3x3 local-max NMS; classes 8,9 keep everything; borders of other classes -> 0
__global__ void nms_kernel(const float* __restrict__ heat, float* __restrict__ hk, int n)
{
    int i = blockIdx.x * blockDim.x + threadIdx.x;
    if (i >= n) return;
    int w = i % WW, h2 = (i / WW) % HH, c = (i / HWD) % NC;
    float v = heat[i];
    float keep;
    if (c >= 8) keep = v;
    else if (h2 == 0 || h2 == HH - 1 || w == 0 || w == WW - 1) keep = 0.f;
    else {
        float mx = -INFINITY;
        for (int dy = -1; dy <= 1; ++dy)
            for (int dx = -1; dx <= 1; ++dx) {
                float u = heat[i + dy * WW + dx];
                mx = mx > u ? mx : u;
            }
        keep = (v == mx) ? v : 0.f;
    }
    hk[i] = keep;
}

// iterative top-200 argmax per batch (clobbers hk)
__global__ __launch_bounds__(1024)
void topk_kernel(float* __restrict__ hk, int* __restrict__ top, int perB)
{
    __shared__ float rv[1024];
    __shared__ int   ri[1024];
    int b = blockIdx.x, tid = threadIdx.x;
    float* base = hk + (size_t)b * perB;
    for (int it = 0; it < NP; ++it) {
        float best = -INFINITY; int bi = 0;
        for (int i = tid; i < perB; i += 1024) {
            float v = base[i];
            if (v > best) { best = v; bi = i; }
        }
        rv[tid] = best; ri[tid] = bi;
        __syncthreads();
        for (int s = 512; s > 0; s >>= 1) {
            if (tid < s) {
                if (rv[tid + s] > rv[tid] ||
                    (rv[tid + s] == rv[tid] && ri[tid + s] < ri[tid])) {
                    rv[tid] = rv[tid + s]; ri[tid] = ri[tid + s];
                }
            }
            __syncthreads();
        }
        if (tid == 0) {
            top[b * NP + it] = ri[0];
            base[ri[0]] = -INFINITY;
        }
        __syncthreads();
    }
}

// t0 = gather(feat) + av[cls]@aq_w^T + aq_b ; qpos from spatial index
__global__ void build_queries(const float* __restrict__ feat, const int* __restrict__ top,
                              const float* __restrict__ av, const float* __restrict__ aqw,
                              const float* __restrict__ aqb, float* __restrict__ t0,
                              float* __restrict__ qpos)
{
    int i = blockIdx.x * blockDim.x + threadIdx.x;
    if (i >= BN * NP * HID) return;
    int c = i % HID, q = (i / HID) % NP, b = i / (HID * NP);
    int idx = top[b * NP + q];
    int cls = idx / HWD, sp = idx % HWD;
    float s = aqb[c];
    for (int k = 0; k < TD; ++k) s += av[cls * TD + k] * aqw[c * TD + k];
    t0[((size_t)(b * NP + q)) * HID + c] =
        feat[((size_t)(b * HID + c)) * HWD + sp] + s;
    if (c == 0) {
        qpos[(b * NP + q) * 2 + 0] = (float)(sp / WW) + 0.5f;
        qpos[(b * NP + q) * 2 + 1] = (float)(sp % WW) + 0.5f;
    }
}

// posembed layer1: relu((pos@w1^T+b1)*g+bb). pos==null -> BEV grid coordinates.
__global__ void pe_l1(const float* __restrict__ pos, const float* __restrict__ w1,
                      const float* __restrict__ b1, const float* __restrict__ g,
                      const float* __restrict__ bb, float* __restrict__ h, int rows)
{
    int i = blockIdx.x * blockDim.x + threadIdx.x;
    if (i >= rows * HID) return;
    int c = i % HID, r = i / HID;
    float p0, p1;
    if (pos) { p0 = pos[r * 2]; p1 = pos[r * 2 + 1]; }
    else { int s = r % HWD; p0 = (float)(s / WW) + 0.5f; p1 = (float)(s % WW) + 0.5f; }
    float v = (p0 * w1[c * 2] + p1 * w1[c * 2 + 1] + b1[c]) * g[c] + bb[c];
    h[i] = v > 0.f ? v : 0.f;
}

// mk[b*HW+s][c] += feat[b][c][s]
__global__ void add_feat_kernel(const float* __restrict__ feat, float* __restrict__ mk, int n)
{
    int i = blockIdx.x * blockDim.x + threadIdx.x;
    if (i >= n) return;
    int c = i % HID, s = (i / HID) % HWD, b = i / (HID * HWD);
    mk[i] += feat[((size_t)(b * HID + c)) * HWD + s];
}

// Y[r][oc] = (X[r]+X2[r]) @ Wt[oc]^T + bias[oc]
__global__ void proj_kernel(const float* __restrict__ X, const float* __restrict__ X2,
                            const float* __restrict__ Wt, const float* __restrict__ bias,
                            float* __restrict__ Y, int rows, int K, int N)
{
    int i = blockIdx.x * blockDim.x + threadIdx.x;
    if (i >= rows * N) return;
    int oc = i % N, r = i / N;
    const float* xp = X + (size_t)r * K;
    const float* x2 = X2 ? X2 + (size_t)r * K : nullptr;
    const float* wp = Wt + (size_t)oc * K;
    float s = bias[oc];
    for (int k = 0; k < K; ++k) s += (xp[k] + (x2 ? x2[k] : 0.f)) * wp[k];
    Y[i] = s;
}

// self-attention over 200 queries; qkv [B][NP][384]; o [B][NP][128]
__global__ void self_attn_kernel(const float* __restrict__ qkv, float* __restrict__ o)
{
    int i = blockIdx.x * blockDim.x + threadIdx.x;
    if (i >= BN * NHD * NP) return;
    int q = i % NP, h = (i / NP) % NHD, b = i / (NP * NHD);
    const float* qp = qkv + ((size_t)(b * NP + q)) * 384 + h * HD;
    float m = -INFINITY, den = 0.f, acc[HD];
    for (int d = 0; d < HD; ++d) acc[d] = 0.f;
    for (int kq = 0; kq < NP; ++kq) {
        const float* kp = qkv + ((size_t)(b * NP + kq)) * 384 + 128 + h * HD;
        const float* vp = kp + 128;
        float s = 0.f;
        for (int d = 0; d < HD; ++d) s += qp[d] * kp[d];
        s *= 0.25f;
        float nm = m > s ? m : s;
        float sc = __expf(m - nm);
        float e  = __expf(s - nm);
        den = den * sc + e;
        for (int d = 0; d < HD; ++d) acc[d] = acc[d] * sc + e * vp[d];
        m = nm;
    }
    float inv = 1.f / den;
    float* op = o + ((size_t)(b * NP + q)) * HID + h * HD;
    for (int d = 0; d < HD; ++d) op[d] = acc[d] * inv;
}

// out = LN(resid + src @ Wo^T + bo)   (one row per block, 128 threads)
__global__ __launch_bounds__(128)
void proj_add_ln(const float* __restrict__ src, const float* __restrict__ Wo,
                 const float* __restrict__ bo, const float* __restrict__ resid,
                 const float* __restrict__ g, const float* __restrict__ beta,
                 float* __restrict__ outp)
{
    __shared__ float red[128];
    int r = blockIdx.x, c = threadIdx.x;
    const float* sp = src + (size_t)r * HID;
    float s = bo[c];
    for (int k = 0; k < HID; ++k) s += sp[k] * Wo[c * HID + k];
    float v = resid[(size_t)r * HID + c] + s;
    red[c] = v; __syncthreads();
    for (int st = 64; st > 0; st >>= 1) { if (c < st) red[c] += red[c + st]; __syncthreads(); }
    float mean = red[0] * (1.f / HID);
    __syncthreads();
    float d = v - mean;
    red[c] = d * d; __syncthreads();
    for (int st = 64; st > 0; st >>= 1) { if (c < st) red[c] += red[c + st]; __syncthreads(); }
    float var = red[0] * (1.f / HID);
    outp[(size_t)r * HID + c] = d * rsqrtf(var + 1e-5f) * g[c] + beta[c];
}

// two-pass softmax cross-attention over HW=32400 keys (L2-resident kh/vh)
__global__ __launch_bounds__(256)
void cross_attn_kernel(const float* __restrict__ qcr, const float* __restrict__ kh,
                       const float* __restrict__ vh, float* __restrict__ o)
{
    __shared__ float red[256];
    int blk = blockIdx.x;
    int q = blk % NP, h = (blk / NP) % NHD, b = blk / (NP * NHD);
    int tid = threadIdx.x;
    float qv[HD];
    const float* qp = qcr + ((size_t)(b * NP + q)) * HID + h * HD;
    for (int d = 0; d < HD; ++d) qv[d] = qp[d];
    const float* kb = kh + ((size_t)b * HWD) * HID + h * HD;
    const float* vb = vh + ((size_t)b * HWD) * HID + h * HD;

    float mx = -INFINITY;
    for (int s = tid; s < HWD; s += 256) {
        const float* kp = kb + (size_t)s * HID;
        float sc = 0.f;
        for (int d = 0; d < HD; ++d) sc += qv[d] * kp[d];
        sc *= 0.25f;
        mx = mx > sc ? mx : sc;
    }
    red[tid] = mx; __syncthreads();
    for (int st = 128; st > 0; st >>= 1) { if (tid < st) red[tid] = fmaxf(red[tid], red[tid + st]); __syncthreads(); }
    float gmax = red[0];
    __syncthreads();

    float den = 0.f, acc[HD];
    for (int d = 0; d < HD; ++d) acc[d] = 0.f;
    for (int s = tid; s < HWD; s += 256) {
        const float* kp = kb + (size_t)s * HID;
        float sc = 0.f;
        for (int d = 0; d < HD; ++d) sc += qv[d] * kp[d];
        float e = __expf(sc * 0.25f - gmax);
        den += e;
        const float* vp = vb + (size_t)s * HID;
        for (int d = 0; d < HD; ++d) acc[d] += e * vp[d];
    }
    red[tid] = den; __syncthreads();
    for (int st = 128; st > 0; st >>= 1) { if (tid < st) red[tid] += red[tid + st]; __syncthreads(); }
    float gden = red[0];
    __syncthreads();

    float* op = o + ((size_t)(b * NP + q)) * HID + h * HD;
    for (int d = 0; d < HD; ++d) {
        red[tid] = acc[d]; __syncthreads();
        for (int st = 128; st > 0; st >>= 1) { if (tid < st) red[tid] += red[tid + st]; __syncthreads(); }
        if (tid == 0) op[d] = red[0] / gden;
        __syncthreads();
    }
}

// out = LN(t + relu(t@w1^T+b1)@w2^T+b2)   (one row per block, 256 threads)
__global__ __launch_bounds__(256)
void ffn_ln_kernel(const float* __restrict__ t, const float* __restrict__ w1,
                   const float* __restrict__ b1, const float* __restrict__ w2,
                   const float* __restrict__ b2, const float* __restrict__ g,
                   const float* __restrict__ beta, float* __restrict__ outp)
{
    __shared__ float hid[FFN];
    __shared__ float red[128];
    __shared__ float xin[HID];
    int r = blockIdx.x, tid = threadIdx.x;
    if (tid < HID) xin[tid] = t[(size_t)r * HID + tid];
    __syncthreads();
    {
        float s = b1[tid];
        for (int k = 0; k < HID; ++k) s += xin[k] * w1[tid * HID + k];
        hid[tid] = s > 0.f ? s : 0.f;
    }
    __syncthreads();
    float myv = 0.f;
    if (tid < HID) {
        float s = b2[tid];
        for (int k = 0; k < FFN; ++k) s += hid[k] * w2[tid * FFN + k];
        myv = xin[tid] + s;
        red[tid] = myv;
    }
    __syncthreads();
    for (int st = 64; st > 0; st >>= 1) { if (tid < st) red[tid] += red[tid + st]; __syncthreads(); }
    float mean = red[0] * (1.f / HID);
    __syncthreads();
    float d = myv - mean;
    if (tid < HID) red[tid] = d * d;
    __syncthreads();
    for (int st = 64; st > 0; st >>= 1) { if (tid < st) red[tid] += red[tid + st]; __syncthreads(); }
    float var = red[0] * (1.f / HID);
    if (tid < HID)
        outp[(size_t)r * HID + tid] = d * rsqrtf(var + 1e-5f) * g[tid] + beta[tid];
}

struct HeadP { const float *w1, *g, *b, *w2, *b2; int oc; int off; };

__global__ __launch_bounds__(64)
void heads_kernel(const float* __restrict__ t, const float* __restrict__ qpos,
                  HeadP h0, HeadP h1, HeadP h2, HeadP h3, HeadP h4, HeadP h5,
                  float* __restrict__ preds)
{
    __shared__ float xin[HID];
    __shared__ float h1s[64];
    int r = blockIdx.x, tid = threadIdx.x;
    int b = r / NP, q = r % NP;
    xin[tid]      = t[(size_t)r * HID + tid];
    xin[tid + 64] = t[(size_t)r * HID + tid + 64];
    __syncthreads();
    HeadP hs[6] = {h0, h1, h2, h3, h4, h5};
    for (int hi = 0; hi < 6; ++hi) {
        HeadP H = hs[hi];
        float s = 0.f;
        for (int k = 0; k < HID; ++k) s += H.w1[tid * HID + k] * xin[k];
        s = s * H.g[tid] + H.b[tid];
        h1s[tid] = s > 0.f ? s : 0.f;
        __syncthreads();
        if (tid < H.oc) {
            float o2 = H.b2[tid];
            for (int k = 0; k < 64; ++k) o2 += H.w2[tid * 64 + k] * h1s[k];
            if (hi == 0) o2 += qpos[r * 2 + tid];
            preds[((size_t)b * 70 + H.off + tid) * NP + q] = o2;
        }
        __syncthreads();
    }
}

// ---------------------------------------------------------------------------
extern "C" void kernel_launch(void* const* d_in, const int* in_sizes, int n_in,
                              void* d_out, int out_size, void* d_ws, size_t ws_size,
                              hipStream_t stream)
{
    (void)in_sizes; (void)n_in; (void)out_size; (void)ws_size;
    const float* X        = (const float*)d_in[0];
    const float* shared_w = (const float*)d_in[1];
    const float* shared_b = (const float*)d_in[2];
    const float* hm_w1    = (const float*)d_in[3];
    const float* hm_g     = (const float*)d_in[4];
    const float* hm_b     = (const float*)d_in[5];
    const float* hm_w2    = (const float*)d_in[6];
    const float* hm_b2    = (const float*)d_in[7];
    const float* aw1      = (const float*)d_in[8];
    const float* ab1      = (const float*)d_in[9];
    const float* aw2      = (const float*)d_in[10];
    const float* ab2      = (const float*)d_in[11];
    const float* msc      = (const float*)d_in[12];
    const float* mbi      = (const float*)d_in[13];
    const float* av       = (const float*)d_in[14];
    const float* aqw      = (const float*)d_in[15];
    const float* aqb      = (const float*)d_in[16];
    const float* sp_w1    = (const float*)d_in[17];
    const float* sp_b1    = (const float*)d_in[18];
    const float* sp_g     = (const float*)d_in[19];
    const float* sp_bb    = (const float*)d_in[20];
    const float* sp_w2    = (const float*)d_in[21];
    const float* sp_b2    = (const float*)d_in[22];
    const float* cp_w1    = (const float*)d_in[23];
    const float* cp_b1    = (const float*)d_in[24];
    const float* cp_g     = (const float*)d_in[25];
    const float* cp_bb    = (const float*)d_in[26];
    const float* cp_w2    = (const float*)d_in[27];
    const float* cp_b2    = (const float*)d_in[28];
    const float* sa_wi    = (const float*)d_in[29];
    const float* sa_bi    = (const float*)d_in[30];
    const float* sa_wo    = (const float*)d_in[31];
    const float* sa_bo    = (const float*)d_in[32];
    const float* ca_wi    = (const float*)d_in[33];
    const float* ca_bi    = (const float*)d_in[34];
    const float* ca_wo    = (const float*)d_in[35];
    const float* ca_bo    = (const float*)d_in[36];
    const float* ln1_g    = (const float*)d_in[37];
    const float* ln1_b    = (const float*)d_in[38];
    const float* ln2_g    = (const float*)d_in[39];
    const float* ln2_b    = (const float*)d_in[40];
    const float* ln3_g    = (const float*)d_in[41];
    const float* ln3_b    = (const float*)d_in[42];
    const float* ff_w1    = (const float*)d_in[43];
    const float* ff_b1    = (const float*)d_in[44];
    const float* ff_w2    = (const float*)d_in[45];
    const float* ff_b2    = (const float*)d_in[46];

    HeadP H0{(const float*)d_in[47], (const float*)d_in[48], (const float*)d_in[49],
             (const float*)d_in[50], (const float*)d_in[51], 2, 0};
    HeadP H1{(const float*)d_in[52], (const float*)d_in[53], (const float*)d_in[54],
             (const float*)d_in[55], (const float*)d_in[56], 1, 2};
    HeadP H2{(const float*)d_in[57], (const float*)d_in[58], (const float*)d_in[59],
             (const float*)d_in[60], (const float*)d_in[61], 3, 3};
    HeadP H3{(const float*)d_in[62], (const float*)d_in[63], (const float*)d_in[64],
             (const float*)d_in[65], (const float*)d_in[66], 2, 6};
    HeadP H4{(const float*)d_in[67], (const float*)d_in[68], (const float*)d_in[69],
             (const float*)d_in[70], (const float*)d_in[71], 2, 8};
    HeadP H5{(const float*)d_in[72], (const float*)d_in[73], (const float*)d_in[74],
             (const float*)d_in[75], (const float*)d_in[76], 60, 10};

    // ---- workspace carve-out ----
    size_t off = 0;
    char* base = (char*)d_ws;
    auto alloc = [&](size_t nfloats) -> float* {
        float* p = (float*)(base + off);
        off = (off + nfloats * sizeof(float) + 255) & ~(size_t)255;
        return p;
    };
    float* feat  = alloc((size_t)BN * HID * HWD);
    float* hbuf  = alloc((size_t)BN * HID * HWD);   // reused as vh later
    float* hmf   = alloc((size_t)BN * TD * HWD);
    float* hmn   = alloc((size_t)BN * TD * HWD);
    float* wdyn  = alloc(NC * TD * 9);
    float* heat  = alloc((size_t)BN * NC * HWD);
    float* heatk = alloc((size_t)BN * NC * HWD);
    int*   top   = (int*)alloc(BN * NP);
    float* t0    = alloc(BN * NP * HID);
    float* qpos  = alloc(BN * NP * 2);
    float* peh   = alloc(BN * NP * HID);
    float* qpe   = alloc(BN * NP * HID);
    float* kpeh  = alloc((size_t)BN * HWD * HID);   // reused as kh later
    float* mk    = alloc((size_t)BN * HWD * HID);
    float* qkv   = alloc(BN * NP * 3 * HID);
    float* attno = alloc(BN * NP * HID);
    float* t1    = alloc(BN * NP * HID);
    float* qcr   = alloc(BN * NP * HID);
    float* t2    = alloc(BN * NP * HID);
    float* t3    = alloc(BN * NP * HID);
    float* kh = kpeh;   // kpe hidden is dead once mk is built
    float* vh = hbuf;   // hm hidden is dead once hm conv2 is done

    float* preds = (float*)d_out;
    float* dense = preds + (size_t)BN * 70 * NP;

    const int WT = (WW + 31) / 32;
    dim3 convG2(BN * HH * WT, 2);
    dim3 convG1(BN * HH * WT, 1);

    // 1) backbone convs (WMMA)
    conv3x3_wmma<<<convG2, 128, 0, stream>>>(X, shared_w, shared_b,
        nullptr, nullptr, nullptr, nullptr, feat, CINX, HID, 0);
    conv3x3_wmma<<<convG2, 128, 0, stream>>>(feat, hm_w1, nullptr,
        hm_g, hm_b, nullptr, nullptr, hbuf, HID, HID, 1);
    conv3x3_wmma<<<convG1, 128, 0, stream>>>(hbuf, hm_w2, hm_b2,
        nullptr, nullptr, nullptr, nullptr, hmf, HID, TD, 0);

    // 2) dynamic anchor conv -> dense output
    wdyn_kernel<<<1, 256, 0, stream>>>(av, aw1, ab1, aw2, ab2, wdyn);
    channel_norm_kernel<<<(BN * HWD + 255) / 256, 256, 0, stream>>>(hmf, hmn, TD, BN * HWD);
    conv3x3_wmma<<<convG1, 128, 0, stream>>>(hmn, wdyn, nullptr,
        nullptr, nullptr, msc, mbi, dense, TD, NC, 2);

    // 3) heatmap NMS + top-k proposals
    int nh = BN * NC * HWD;
    sigmoid_kernel<<<(nh + 255) / 256, 256, 0, stream>>>(dense, heat, nh);
    nms_kernel<<<(nh + 255) / 256, 256, 0, stream>>>(heat, heatk, nh);
    topk_kernel<<<BN, 1024, 0, stream>>>(heatk, top, NC * HWD);

    // 4) queries + positional embeddings
    build_queries<<<(BN * NP * HID + 255) / 256, 256, 0, stream>>>(
        feat, top, av, aqw, aqb, t0, qpos);
    pe_l1<<<(BN * NP * HID + 255) / 256, 256, 0, stream>>>(
        qpos, sp_w1, sp_b1, sp_g, sp_bb, peh, BN * NP);
    gemm_nt_wmma<<<dim3((BN * NP + 31) / 32, 2), 128, 0, stream>>>(
        peh, sp_w2, sp_b2, qpe, BN * NP, HID, HID, 0);
    pe_l1<<<((size_t)BN * HWD * HID + 255) / 256, 256, 0, stream>>>(
        nullptr, cp_w1, cp_b1, cp_g, cp_bb, kpeh, BN * HWD);
    gemm_nt_wmma<<<dim3((BN * HWD + 31) / 32, 2), 128, 0, stream>>>(
        kpeh, cp_w2, cp_b2, mk, BN * HWD, HID, HID, 0);
    add_feat_kernel<<<((size_t)BN * HWD * HID + 255) / 256, 256, 0, stream>>>(
        feat, mk, BN * HWD * HID);

    // 5) self-attention + LN1
    proj_kernel<<<(BN * NP * 384 + 255) / 256, 256, 0, stream>>>(
        t0, qpe, sa_wi, sa_bi, qkv, BN * NP, HID, 3 * HID);
    self_attn_kernel<<<(BN * NHD * NP + 255) / 256, 256, 0, stream>>>(qkv, attno);
    proj_add_ln<<<BN * NP, 128, 0, stream>>>(attno, sa_wo, sa_bo, t0, ln1_g, ln1_b, t1);

    // 6) cross-attention (WMMA K/V projections) + LN2
    gemm_nt_wmma<<<dim3((BN * HWD + 31) / 32, 2), 128, 0, stream>>>(
        mk, ca_wi + (size_t)HID * HID, ca_bi + HID, kh, BN * HWD, HID, HID, 0);
    gemm_nt_wmma<<<dim3((BN * HWD + 31) / 32, 2), 128, 0, stream>>>(
        mk, ca_wi + (size_t)2 * HID * HID, ca_bi + 2 * HID, vh, BN * HWD, HID, HID, 0);
    proj_kernel<<<(BN * NP * HID + 255) / 256, 256, 0, stream>>>(
        t1, qpe, ca_wi, ca_bi, qcr, BN * NP, HID, HID);
    cross_attn_kernel<<<BN * NHD * NP, 256, 0, stream>>>(qcr, kh, vh, attno);
    proj_add_ln<<<BN * NP, 128, 0, stream>>>(attno, ca_wo, ca_bo, t1, ln2_g, ln2_b, t2);

    // 7) FFN + LN3, prediction heads
    ffn_ln_kernel<<<BN * NP, 256, 0, stream>>>(t2, ff_w1, ff_b1, ff_w2, ff_b2,
                                               ln3_g, ln3_b, t3);
    heads_kernel<<<BN * NP, 64, 0, stream>>>(t3, qpos, H0, H1, H2, H3, H4, H5, preds);
}
